// Qwen3VisionAttention_51848845197677
// MI455X (gfx1250) — matrix operand order repaired and checked by reading, not compiled, not run on hardware
//
#include <hip/hip_runtime.h>

// ---------------- problem constants ----------------
#define EMBED   1280
#define NHEADS  16
#define HDIM    80
#define HPAD    96          // HDIM padded to multiple of 32 for WMMA K-steps
#define ROT     40
#define NSEQS   8
#define SLEN    2048
#define TOTAL   (NSEQS * SLEN)   // 16384
#define QKV_N   (3 * EMBED)      // 3840
#define KB      64               // attention key-block

typedef __attribute__((ext_vector_type(16))) __bf16 v16bf;
typedef __attribute__((ext_vector_type(8)))  float  v8f;

__device__ __forceinline__ v8f wmma_bf16(v16bf a, v16bf b, v8f c) {
  return __builtin_amdgcn_wmma_f32_16x16x32_bf16(
      false, a, false, b, (short)0, c, false, false);
}

union Frag { v16bf v; uint4 u[2]; };

// A-operand fragment (16x32, MxK): lane L<16: row M=L, K={0..7,16..23};
// lane L>=16: row M=L-16, K={8..15,24..31}.
__device__ __forceinline__ v16bf load_frag_a(const __bf16* base, int stride_elems) {
  int lane = threadIdx.x & 31;
  const __bf16* p = base + (size_t)(lane & 15) * stride_elems + ((lane >> 4) << 3);
  Frag f;
  f.u[0] = *(const uint4*)(p);
  f.u[1] = *(const uint4*)(p + 16);
  return f.v;
}

// B-operand fragment (32x16, KxN) from [N][K] staging:
// lane L<16: col N=L, K={0..7,16..23}; lane L>=16: col N=L-16, K={8..15,24..31}.
__device__ __forceinline__ v16bf load_frag_b(const __bf16* baseNT, int stride_elems) {
  int lane = threadIdx.x & 31;
  const __bf16* p = baseNT + (size_t)(lane & 15) * stride_elems + ((lane >> 4) << 3);
  Frag f;
  f.u[0] = *(const uint4*)(p);
  f.u[1] = *(const uint4*)(p + 16);
  return f.v;
}

// ---- CDNA5 async global->LDS DMA (ASYNCcnt-tracked) ----
__device__ __forceinline__ void async_load_b128(unsigned lds_addr, const void* gaddr) {
  asm volatile("global_load_async_to_lds_b128 %0, %1, off"
               :: "v"(lds_addr), "v"(gaddr) : "memory");
}
// async done-tokens retire in issue order: waiting <=N leaves the N newest in flight
__device__ __forceinline__ void wait_async_le0() { asm volatile("s_wait_asynccnt 0" ::: "memory"); }
__device__ __forceinline__ void wait_async_le4() { asm volatile("s_wait_asynccnt 4" ::: "memory"); }
__device__ __forceinline__ void wait_async_le6() { asm volatile("s_wait_asynccnt 6" ::: "memory"); }

// ---------------- stage 0: precision conversion ----------------
__global__ __launch_bounds__(256) void cvt_f32_to_bf16(const float* __restrict__ in,
                                                       __bf16* __restrict__ out, int n) {
  int i = blockIdx.x * 256 + threadIdx.x;
  if (i < n) out[i] = (__bf16)in[i];
}

__global__ __launch_bounds__(256) void cvt_transpose_bf16(const float* __restrict__ in,
                                                          __bf16* __restrict__ out,
                                                          int R, int C) {
  int i = blockIdx.x * 256 + threadIdx.x;
  if (i >= R * C) return;
  int r = i / C, c = i % C;
  out[(size_t)c * R + r] = (__bf16)in[i];
}

// ---------------- LDS-tiled bf16 WMMA GEMM: Out[M,N] = A[M,K]*Bt[N,K]^T + bias ----------
// Workgroup macro-tile 128x128, BK=32, async-DMA double-buffered LDS (32 KB).
// 8 waves in 4x2: each wave owns 32x64 (2x4 sub-tiles), 8 WMMAs + 12 ds_load_b128 / step.
template <bool OUT_BF16>
__global__ __launch_bounds__(256) void gemm_lds_tile(const __bf16* __restrict__ A,
                                                     const __bf16* __restrict__ Bt,
                                                     const float* __restrict__ bias,
                                                     void* __restrict__ Out,
                                                     int M, int N, int K) {
  __shared__ __bf16 Alds[2][128 * 32];   // 2 x 8 KB
  __shared__ __bf16 Blds[2][128 * 32];   // 2 x 8 KB

  const int tid  = threadIdx.x;
  const int w    = tid >> 5, lane = tid & 31;
  const int nbn  = N >> 7;
  const int m0   = (blockIdx.x / nbn) << 7;
  const int n0   = (blockIdx.x % nbn) << 7;
  const int wrow = w & 3, wcol = w >> 2;

  const char* Ab = (const char*)A;
  const char* Bb = (const char*)Bt;
  const unsigned alds0 = (unsigned)(size_t)&Alds[0][0];
  const unsigned blds0 = (unsigned)(size_t)&Blds[0][0];

  // 512 A chunks + 512 B chunks of 16 B per stage -> exactly 4 async ops per thread
  auto stage = [&](int buf, int k0) {
    const unsigned al = alds0 + buf * (128 * 32 * 2);
    const unsigned bl = blds0 + buf * (128 * 32 * 2);
#pragma unroll
    for (int i = 0; i < 2; ++i) {
      int c = tid + i * 256;
      int row = c >> 2, cc = (c & 3) << 4;
      async_load_b128(al + c * 16, Ab + ((size_t)(m0 + row) * K + k0) * 2 + cc);
    }
#pragma unroll
    for (int i = 0; i < 2; ++i) {
      int c = tid + i * 256;
      int row = c >> 2, cc = (c & 3) << 4;
      async_load_b128(bl + c * 16, Bb + ((size_t)(n0 + row) * K + k0) * 2 + cc);
    }
  };

  v8f acc[2][4];
#pragma unroll
  for (int mi = 0; mi < 2; ++mi)
#pragma unroll
    for (int ni = 0; ni < 4; ++ni) acc[mi][ni] = v8f{};

  const int NK = K >> 5;
  stage(0, 0);
  for (int i = 0; i < NK; ++i) {
    const int cur = i & 1;
    if (i + 1 < NK) { stage(cur ^ 1, (i + 1) << 5); wait_async_le4(); }
    else            { wait_async_le0(); }
    __syncthreads();

    const __bf16* Asb = &Alds[cur][0];
    const __bf16* Bsb = &Blds[cur][0];
    v16bf af0 = load_frag_a(Asb + (wrow * 32) * 32, 32);
    v16bf af1 = load_frag_a(Asb + (wrow * 32 + 16) * 32, 32);
#pragma unroll
    for (int ni = 0; ni < 4; ++ni) {
      v16bf b = load_frag_b(Bsb + (wcol * 64 + ni * 16) * 32, 32);
      acc[0][ni] = wmma_bf16(af0, b, acc[0][ni]);
      acc[1][ni] = wmma_bf16(af1, b, acc[1][ni]);
    }
    __syncthreads();   // all reads of `cur` done before it is re-staged
  }

  const int cn = lane & 15, rbse = (lane >> 4) << 3;
#pragma unroll
  for (int mi = 0; mi < 2; ++mi)
#pragma unroll
    for (int ni = 0; ni < 4; ++ni) {
      const int col = n0 + wcol * 64 + ni * 16 + cn;
      const float bv = bias[col];
#pragma unroll
      for (int r = 0; r < 8; ++r) {
        const int row = m0 + wrow * 32 + mi * 16 + rbse + r;
        const float val = acc[mi][ni][r] + bv;
        if (OUT_BF16) ((__bf16*)Out)[(size_t)row * N + col] = (__bf16)val;
        else          ((float*)Out)[(size_t)row * N + col] = val;
      }
    }
}

// ---------------- stage 2: RoPE + scatter into attention layouts ----------------
__global__ __launch_bounds__(256) void rope_scatter(const __bf16* __restrict__ qkv,
                                                    const float* __restrict__ cosT,
                                                    const float* __restrict__ sinT,
                                                    __bf16* __restrict__ qp,
                                                    __bf16* __restrict__ kp,
                                                    __bf16* __restrict__ vt) {
  int idx = blockIdx.x * 256 + threadIdx.x;
  if (idx >= TOTAL * EMBED) return;
  int t  = idx / EMBED;
  int hd = idx - t * EMBED;
  int h  = hd / HDIM, d = hd - h * HDIM;
  int s  = t >> 11, l = t & (SLEN - 1);

  size_t rowq = (size_t)t * QKV_N + hd;
  float qv = (float)qkv[rowq];
  float kv = (float)qkv[rowq + EMBED];
  float vv = (float)qkv[rowq + 2 * EMBED];

  int d2 = (d < ROT) ? d + ROT : d - ROT;
  float sgn = (d < ROT) ? -1.f : 1.f;
  size_t rowq2 = (size_t)t * QKV_N + h * HDIM + d2;
  float q2 = (float)qkv[rowq2];
  float k2 = (float)qkv[rowq2 + EMBED];

  int dm = (d < ROT) ? d : d - ROT;
  float c  = cosT[t * ROT + dm];
  float sn = sinT[t * ROT + dm];

  float qo = qv * c + sgn * q2 * sn;
  float ko = kv * c + sgn * k2 * sn;

  size_t sh   = (size_t)(s * NHEADS + h);
  size_t qrow = (sh * SLEN + l) * HPAD;
  qp[qrow + d] = (__bf16)qo;
  kp[qrow + d] = (__bf16)ko;
  if (d < HPAD - HDIM) {
    qp[qrow + HDIM + d] = (__bf16)0.f;
    kp[qrow + HDIM + d] = (__bf16)0.f;
  }
  vt[(sh * HDIM + d) * SLEN + l] = (__bf16)vv;
}

// ---------------- stage 3: flash attention ----------------
// grid: (s*NHEADS)*(SLEN/128); 256 thr = 8 waves; wave = 16 query rows.
// K (64x96) and V (80x64) tiles async-DMA'd into double-buffered LDS; block i+1's
// DMA overlaps block i's QK^T/softmax/PV. LDS: 24+20+16 = 60 KB.
__global__ __launch_bounds__(256) void flash_attn(const __bf16* __restrict__ qp,
                                                  const __bf16* __restrict__ kp,
                                                  const __bf16* __restrict__ vt,
                                                  __bf16* __restrict__ ao) {
  __shared__ __bf16 Klds[2][KB * HPAD];       // 2 x 12 KB
  __shared__ __bf16 Vlds[2][HDIM * KB];       // 2 x 10 KB
  __shared__ __bf16 Plds[8][16 * KB];         // 16 KB (per-wave P staging)

  const int w    = threadIdx.x >> 5;
  const int lane = threadIdx.x & 31;
  const int tid  = threadIdx.x;
  const int qb   = blockIdx.x & 15;
  const int sh   = blockIdx.x >> 4;

  const __bf16* qbase = qp + ((size_t)sh * SLEN + qb * 128 + w * 16) * HPAD;
  const char*   kbyte = (const char*)(kp + (size_t)sh * SLEN * HPAD);
  const char*   vbyte = (const char*)(vt + (size_t)sh * HDIM * SLEN);
  const unsigned klds0 = (unsigned)(size_t)&Klds[0][0];
  const unsigned vlds0 = (unsigned)(size_t)&Vlds[0][0];

  // K tile: rows contiguous in both global and LDS -> flat 768-chunk copy.
  // V tile: 640 chunks (80 rows x 8); tail threads wrap onto already-covered
  // chunks (idempotent) so every thread issues exactly 6 async ops per stage.
  auto stage = [&](int buf, int kb) {
    const unsigned kl = klds0 + buf * (KB * HPAD * 2);
    const unsigned vl = vlds0 + buf * (HDIM * KB * 2);
#pragma unroll
    for (int i = 0; i < 3; ++i) {                 // 3*256 = 768 K chunks
      int c = tid + i * 256;
      async_load_b128(kl + c * 16, kbyte + (size_t)kb * (HPAD * 2) + c * 16);
    }
#pragma unroll
    for (int i = 0; i < 3; ++i) {                 // 640 V chunks (+dup tail)
      int c = tid + i * 256;
      if (c >= 640) c -= 128;
      int d = c >> 3, cc = (c & 7) << 4;
      async_load_b128(vl + c * 16, vbyte + ((size_t)d * SLEN + kb) * 2 + cc);
    }
  };

  v16bf qf[3];
#pragma unroll
  for (int kk = 0; kk < 3; ++kk) qf[kk] = load_frag_a(qbase + kk * 32, HPAD);

  v8f oacc[5] = {v8f{}, v8f{}, v8f{}, v8f{}, v8f{}};
  float mrow[8], lrow[8];
#pragma unroll
  for (int r = 0; r < 8; ++r) { mrow[r] = -1e30f; lrow[r] = 0.f; }

  const float scale = 0.1118033988749895f;   // 1/sqrt(80)
  __bf16* myP = &Plds[w][0];
  const int cn = lane & 15;
  const int rb = (lane >> 4) << 3;
  const int NB = SLEN / KB;                   // 32

  stage(0, 0);
  for (int i = 0; i < NB; ++i) {
    const int cur = i & 1;
    if (i + 1 < NB) { stage(cur ^ 1, (i + 1) * KB); wait_async_le6(); }
    else            { wait_async_le0(); }
    __syncthreads();

    const __bf16* Kt = &Klds[cur][0];
    const __bf16* Vt = &Vlds[cur][0];

    // ---- S = Q K^T : 4 N-tiles x 3 K-steps, all K-frags from LDS ----
    v8f sacc[4];
#pragma unroll
    for (int nt = 0; nt < 4; ++nt) {
      v8f s = v8f{};
#pragma unroll
      for (int kk = 0; kk < 3; ++kk)
        s = wmma_bf16(qf[kk], load_frag_b(Kt + (nt * 16) * HPAD + kk * 32, HPAD), s);
      sacc[nt] = s;
    }

    // ---- online softmax ----
#pragma unroll
    for (int r = 0; r < 8; ++r) {
      float mx = -1e30f;
#pragma unroll
      for (int nt = 0; nt < 4; ++nt) {
        sacc[nt][r] *= scale;
        mx = fmaxf(mx, sacc[nt][r]);
      }
      for (int off = 1; off < 16; off <<= 1) mx = fmaxf(mx, __shfl_xor(mx, off, 32));
      float nm = fmaxf(mrow[r], mx);
      float alpha = __expf(mrow[r] - nm);
      mrow[r] = nm;
      lrow[r] *= alpha;
#pragma unroll
      for (int nt = 0; nt < 5; ++nt) oacc[nt][r] *= alpha;
      float sum = 0.f;
#pragma unroll
      for (int nt = 0; nt < 4; ++nt) {
        float p = __expf(sacc[nt][r] - nm);
        sacc[nt][r] = p;
        sum += p;
      }
      for (int off = 1; off < 16; off <<= 1) sum += __shfl_xor(sum, off, 32);
      lrow[r] += sum;
    }

    // ---- C-layout -> A-layout via per-wave LDS staging ----
#pragma unroll
    for (int nt = 0; nt < 4; ++nt)
#pragma unroll
      for (int r = 0; r < 8; ++r)
        myP[(rb + r) * KB + nt * 16 + cn] = (__bf16)sacc[nt][r];

    // ---- O += P V : 2 K-steps x 5 D-tiles, operands in LDS ----
#pragma unroll
    for (int kk = 0; kk < 2; ++kk) {
      v16bf pf = load_frag_a(myP + kk * 32, KB);
#pragma unroll
      for (int nt = 0; nt < 5; ++nt)
        oacc[nt] = wmma_bf16(pf, load_frag_b(Vt + (nt * 16) * KB + kk * 32, KB), oacc[nt]);
    }
    __syncthreads();   // all reads of `cur` done before it is re-staged
  }

  const int s_ = sh >> 4, h_ = sh & 15;
#pragma unroll
  for (int nt = 0; nt < 5; ++nt)
#pragma unroll
    for (int r = 0; r < 8; ++r) {
      int row = qb * 128 + w * 16 + rb + r;
      size_t t = (size_t)s_ * SLEN + row;
      ao[t * EMBED + h_ * HDIM + nt * 16 + cn] = (__bf16)(oacc[nt][r] / lrow[r]);
    }
}

// ---------------- host orchestration ----------------
extern "C" void kernel_launch(void* const* d_in, const int* in_sizes, int n_in,
                              void* d_out, int out_size, void* d_ws, size_t ws_size,
                              hipStream_t stream) {
  const float* x    = (const float*)d_in[0];
  const float* cosT = (const float*)d_in[2];
  const float* sinT = (const float*)d_in[3];
  const float* Wqkv = (const float*)d_in[4];
  const float* bqkv = (const float*)d_in[5];
  const float* Wout = (const float*)d_in[6];
  const float* bout = (const float*)d_in[7];

  char* w = (char*)d_ws;
  size_t off = 0;
  __bf16* xb     = (__bf16*)(w + off); off += (size_t)TOTAL * EMBED * 2;
  __bf16* wqkvT  = (__bf16*)(w + off); off += (size_t)QKV_N * EMBED * 2;
  __bf16* woutT  = (__bf16*)(w + off); off += (size_t)EMBED * EMBED * 2;
  __bf16* qkv_s  = (__bf16*)(w + off); off += (size_t)TOTAL * QKV_N * 2;
  __bf16* qp     = (__bf16*)(w + off); off += (size_t)NSEQS * NHEADS * SLEN * HPAD * 2;
  __bf16* kp     = (__bf16*)(w + off); off += (size_t)NSEQS * NHEADS * SLEN * HPAD * 2;
  __bf16* vt     = (__bf16*)(w + off); off += (size_t)NSEQS * NHEADS * HDIM * SLEN * 2;
  __bf16* ao     = qkv_s;  // reuse: qkv_s fully consumed by rope_scatter before attn

  {
    int n = TOTAL * EMBED;
    cvt_f32_to_bf16<<<(n + 255) / 256, 256, 0, stream>>>(x, xb, n);
    int nw = EMBED * QKV_N;
    cvt_transpose_bf16<<<(nw + 255) / 256, 256, 0, stream>>>(Wqkv, wqkvT, EMBED, QKV_N);
    int no = EMBED * EMBED;
    cvt_transpose_bf16<<<(no + 255) / 256, 256, 0, stream>>>(Wout, woutT, EMBED, EMBED);
  }

  gemm_lds_tile<true><<<(TOTAL / 128) * (QKV_N / 128), 256, 0, stream>>>(
      xb, wqkvT, bqkv, (void*)qkv_s, TOTAL, QKV_N, EMBED);

  {
    int n = TOTAL * EMBED;
    rope_scatter<<<(n + 255) / 256, 256, 0, stream>>>(qkv_s, cosT, sinT, qp, kp, vt);
  }

  flash_attn<<<NSEQS * NHEADS * (SLEN / 128), 256, 0, stream>>>(qp, kp, vt, ao);

  gemm_lds_tile<false><<<(TOTAL / 128) * (EMBED / 128), 256, 0, stream>>>(
      ao, woutT, bout, d_out, TOTAL, EMBED, EMBED);
}
